// GAT_18777597018723
// MI455X (gfx1250) — compile-verified, matrix operand order checked by reading, not run on hardware
//
#include <hip/hip_runtime.h>
#include <hip/hip_bf16.h>
#include <cstdint>

#define N_NODES 50000
#define N_EDGES 600000
#define DIN     256
#define HDIM    128
#define MIDD    512
#define DOUT    128
#define NLAYERS 2
#define LN_EPS  1e-12f

typedef __attribute__((ext_vector_type(16))) _Float16 v16h;
typedef __attribute__((ext_vector_type(8)))  float    v8f;

__device__ __forceinline__ float gelu_exact(float x) {
    return 0.5f * x * (1.0f + erff(x * 0.70710678118654752f));
}
// order-preserving float<->uint encoding for atomicMax on floats
__device__ __forceinline__ unsigned encF(float f) {
    unsigned u = __float_as_uint(f);
    return (u & 0x80000000u) ? ~u : (u | 0x80000000u);
}
__device__ __forceinline__ float decF(unsigned e) {
    return (e & 0x80000000u) ? __uint_as_float(e ^ 0x80000000u)
                             : __uint_as_float(~e);
}

// ---------------------------------------------------------------------------
// Generic fp32 GEMM via f16 WMMA: C[nrows,M] = act(A[nrows,K] @ B[K,M] + bias)
// Block tile 128x128, 8 waves, each wave: 4x2 tiles of v_wmma_f32_16x16x32_f16
// ACT: 0=none 1=relu 2=gelu
// ---------------------------------------------------------------------------
template <int K, int ACT>
__global__ __launch_bounds__(256) void gemm_f16wmma(
    const float* __restrict__ A, const float* __restrict__ B,
    const float* __restrict__ bias, float* __restrict__ C,
    int nrows, int M)
{
    __shared__ _Float16 As[128][32];   // A rows (f16)
    __shared__ _Float16 Bs[128][32];   // B transposed: Bs[n][k]

    const int row0 = blockIdx.x * 128;
    const int col0 = blockIdx.y * 128;
    const int tid  = threadIdx.x;
    const int lane = tid & 31;
    const int wave = tid >> 5;
    const int wm   = (wave >> 2) * 64;   // wave row offset (0 / 64)
    const int wn   = (wave & 3) * 32;    // wave col offset (0..96)
    const int hi   = lane >> 4;
    const int l15  = lane & 15;

    v8f acc[4][2];
#pragma unroll
    for (int mi = 0; mi < 4; ++mi)
#pragma unroll
        for (int ni = 0; ni < 2; ++ni)
#pragma unroll
            for (int v = 0; v < 8; ++v) acc[mi][ni][v] = 0.0f;

    const int arow = tid >> 1;            // 128 rows, 2 threads/row
    const int ahof = (tid & 1) * 16;      // 16 halves each
    const int bk   = tid >> 3;            // 32 k values, 8 threads/k
    const int bnof = (tid & 7) * 16;      // 16 cols each

    for (int k0 = 0; k0 < K; k0 += 32) {
        // ---- fill A tile (fp32 -> f16) ----
        {
            const int gr = row0 + arow;
            if (gr < nrows) {
                const float* ap = A + (size_t)gr * K + k0 + ahof;
#pragma unroll
                for (int j = 0; j < 16; j += 4) {
                    float4 v = *(const float4*)(ap + j);
                    As[arow][ahof + j + 0] = (_Float16)v.x;
                    As[arow][ahof + j + 1] = (_Float16)v.y;
                    As[arow][ahof + j + 2] = (_Float16)v.z;
                    As[arow][ahof + j + 3] = (_Float16)v.w;
                }
            } else {
#pragma unroll
                for (int j = 0; j < 16; ++j) As[arow][ahof + j] = (_Float16)0.0f;
            }
        }
        // ---- fill B tile transposed (coalesced global, scattered LDS) ----
        {
            const float* bp = B + (size_t)(k0 + bk) * M + col0 + bnof;
#pragma unroll
            for (int j = 0; j < 16; j += 4) {
                float4 v = *(const float4*)(bp + j);
                Bs[bnof + j + 0][bk] = (_Float16)v.x;
                Bs[bnof + j + 1][bk] = (_Float16)v.y;
                Bs[bnof + j + 2][bk] = (_Float16)v.z;
                Bs[bnof + j + 3][bk] = (_Float16)v.w;
            }
        }
        __syncthreads();

        // ---- fragments per ISA 16-bit 16x32 layout ----
        v16h bfrag[2];
#pragma unroll
        for (int ni = 0; ni < 2; ++ni) {
            const int nc = wn + 16 * ni + l15;
#pragma unroll
            for (int j = 0; j < 8; ++j) {
                const int ks = 2 * (j & 3) + (j >> 2) * 16 + hi * 8;
                bfrag[ni][2 * j]     = Bs[nc][ks];
                bfrag[ni][2 * j + 1] = Bs[nc][ks + 1];
            }
        }
#pragma unroll
        for (int mi = 0; mi < 4; ++mi) {
            const int mr = wm + 16 * mi + l15;
            v16h afrag;
#pragma unroll
            for (int j = 0; j < 8; ++j) {
                const int ks = 2 * (j & 3) + (j >> 2) * 16 + hi * 8;
                afrag[2 * j]     = As[mr][ks];
                afrag[2 * j + 1] = As[mr][ks + 1];
            }
#pragma unroll
            for (int ni = 0; ni < 2; ++ni) {
                acc[mi][ni] = __builtin_amdgcn_wmma_f32_16x16x32_f16(
                    false, afrag, false, bfrag[ni], (short)0, acc[mi][ni],
                    false, false);
            }
        }
        __syncthreads();
    }

    // ---- epilogue: bias + activation ----
#pragma unroll
    for (int mi = 0; mi < 4; ++mi) {
#pragma unroll
        for (int ni = 0; ni < 2; ++ni) {
            const int n  = col0 + wn + 16 * ni + l15;
            const float bv = bias ? bias[n] : 0.0f;
#pragma unroll
            for (int v = 0; v < 8; ++v) {
                const int m = row0 + wm + 16 * mi + v + hi * 8;
                if (m < nrows) {
                    float x = acc[mi][ni][v] + bv;
                    if (ACT == 1) x = x > 0.0f ? x : 0.0f;
                    else if (ACT == 2) x = gelu_exact(x);
                    C[(size_t)m * M + n] = x;
                }
            }
        }
    }
}

// ---------------------------------------------------------------------------
// alpha_s = hW @ a_src ; alpha_d = hW @ a_dst   (one wave per node)
// ---------------------------------------------------------------------------
__global__ __launch_bounds__(256) void alpha_kernel(
    const float* __restrict__ hW, const float* __restrict__ a_src,
    const float* __restrict__ a_dst, float* __restrict__ as_,
    float* __restrict__ ad_, int n)
{
    const int node = blockIdx.x * 8 + (threadIdx.x >> 5);
    const int lane = threadIdx.x & 31;
    if (node >= n) return;
    float4 v = ((const float4*)(hW + (size_t)node * HDIM))[lane];
    float4 s = ((const float4*)a_src)[lane];
    float4 d = ((const float4*)a_dst)[lane];
    float ss = v.x * s.x + v.y * s.y + v.z * s.z + v.w * s.w;
    float dd = v.x * d.x + v.y * d.y + v.z * d.z + v.w * d.w;
#pragma unroll
    for (int off = 16; off; off >>= 1) {
        ss += __shfl_xor(ss, off);
        dd += __shfl_xor(dd, off);
    }
    if (lane == 0) { as_[node] = ss; ad_[node] = dd; }
}

__global__ void init_kernel(float* __restrict__ agg, unsigned* __restrict__ mE,
                            float* __restrict__ z, int n)
{
    const int i = blockIdx.x * 256 + threadIdx.x;
    if (i < n * HDIM) agg[i] = 0.0f;
    if (i < n) { mE[i] = 0u; z[i] = 0.0f; }
}

__device__ __forceinline__ void edge_sd(const long long* __restrict__ ei,
                                        int eid, int& s, int& d)
{
    if (eid < N_EDGES) { s = (int)ei[eid]; d = (int)ei[N_EDGES + eid]; }
    else               { s = d = eid - N_EDGES; }
}

__global__ void edge_pass1(const long long* __restrict__ ei,
                           const float* __restrict__ as_,
                           const float* __restrict__ ad_,
                           float* __restrict__ ebuf,
                           unsigned* __restrict__ mE, int total)
{
    const int eid = blockIdx.x * 256 + threadIdx.x;
    if (eid >= total) return;
    int s, d; edge_sd(ei, eid, s, d);
    float e = as_[s] + ad_[d];
    e = e > 0.0f ? e : 0.2f * e;          // leaky_relu(0.2)
    ebuf[eid] = e;
    atomicMax(&mE[d], encF(e));
}

__global__ void edge_pass2(const long long* __restrict__ ei,
                           float* __restrict__ ebuf,
                           const unsigned* __restrict__ mE,
                           float* __restrict__ z, int total)
{
    const int eid = blockIdx.x * 256 + threadIdx.x;
    if (eid >= total) return;
    int s, d; edge_sd(ei, eid, s, d);
    float ee = __expf(ebuf[eid] - decF(mE[d]));
    ebuf[eid] = ee;
    atomicAdd(&z[d], ee);
}

// one wave per edge: agg[dst] += alpha * hW[src]
__global__ __launch_bounds__(256) void edge_pass3(
    const long long* __restrict__ ei, const float* __restrict__ ebuf,
    const float* __restrict__ z, const float* __restrict__ hW,
    float* __restrict__ agg, int total)
{
    const int eid  = blockIdx.x * 8 + (threadIdx.x >> 5);
    const int lane = threadIdx.x & 31;
    if (eid >= total) return;
    int s, d; edge_sd(ei, eid, s, d);
    const float alpha = ebuf[eid] / (z[d] + 1e-16f);
    float4 v = ((const float4*)(hW + (size_t)s * HDIM))[lane];
    float* ag = agg + (size_t)d * HDIM + lane * 4;
    atomicAdd(ag + 0, alpha * v.x);
    atomicAdd(ag + 1, alpha * v.y);
    atomicAdd(ag + 2, alpha * v.z);
    atomicAdd(ag + 3, alpha * v.w);
}

// h = LayerNorm(pre_x + gelu(agg + gat_b))   (one wave per node)
__global__ __launch_bounds__(256) void post_gat(
    const float* __restrict__ agg, const float* __restrict__ prex,
    const float* __restrict__ gbias, const float* __restrict__ g,
    const float* __restrict__ b, float* __restrict__ out, int n)
{
    const int node = blockIdx.x * 8 + (threadIdx.x >> 5);
    const int lane = threadIdx.x & 31;
    if (node >= n) return;
    float4 a  = ((const float4*)(agg  + (size_t)node * HDIM))[lane];
    float4 bb = ((const float4*)gbias)[lane];
    float4 p  = ((const float4*)(prex + (size_t)node * HDIM))[lane];
    float4 t;
    t.x = p.x + gelu_exact(a.x + bb.x);
    t.y = p.y + gelu_exact(a.y + bb.y);
    t.z = p.z + gelu_exact(a.z + bb.z);
    t.w = p.w + gelu_exact(a.w + bb.w);
    float sum = t.x + t.y + t.z + t.w;
#pragma unroll
    for (int off = 16; off; off >>= 1) sum += __shfl_xor(sum, off);
    const float mu = sum * (1.0f / HDIM);
    float4 dv = {t.x - mu, t.y - mu, t.z - mu, t.w - mu};
    float vs = dv.x * dv.x + dv.y * dv.y + dv.z * dv.z + dv.w * dv.w;
#pragma unroll
    for (int off = 16; off; off >>= 1) vs += __shfl_xor(vs, off);
    const float inv = rsqrtf(vs * (1.0f / HDIM) + LN_EPS);
    float4 gg = ((const float4*)g)[lane];
    float4 bl = ((const float4*)b)[lane];
    float4 o;
    o.x = dv.x * inv * gg.x + bl.x;
    o.y = dv.y * inv * gg.y + bl.y;
    o.z = dv.z * inv * gg.z + bl.z;
    o.w = dv.w * inv * gg.w + bl.w;
    ((float4*)(out + (size_t)node * HDIM))[lane] = o;
}

// out = LayerNorm(x + y)   (one wave per node; safe in-place on x)
__global__ __launch_bounds__(256) void ln_add(
    const float* __restrict__ x, const float* __restrict__ y,
    const float* __restrict__ g, const float* __restrict__ b,
    float* __restrict__ out, int n)
{
    const int node = blockIdx.x * 8 + (threadIdx.x >> 5);
    const int lane = threadIdx.x & 31;
    if (node >= n) return;
    float4 xa = ((const float4*)(x + (size_t)node * HDIM))[lane];
    float4 ya = ((const float4*)(y + (size_t)node * HDIM))[lane];
    float4 t = {xa.x + ya.x, xa.y + ya.y, xa.z + ya.z, xa.w + ya.w};
    float sum = t.x + t.y + t.z + t.w;
#pragma unroll
    for (int off = 16; off; off >>= 1) sum += __shfl_xor(sum, off);
    const float mu = sum * (1.0f / HDIM);
    float4 dv = {t.x - mu, t.y - mu, t.z - mu, t.w - mu};
    float vs = dv.x * dv.x + dv.y * dv.y + dv.z * dv.z + dv.w * dv.w;
#pragma unroll
    for (int off = 16; off; off >>= 1) vs += __shfl_xor(vs, off);
    const float inv = rsqrtf(vs * (1.0f / HDIM) + LN_EPS);
    float4 gg = ((const float4*)g)[lane];
    float4 bl = ((const float4*)b)[lane];
    float4 o;
    o.x = dv.x * inv * gg.x + bl.x;
    o.y = dv.y * inv * gg.y + bl.y;
    o.z = dv.z * inv * gg.z + bl.z;
    o.w = dv.w * inv * gg.w + bl.w;
    ((float4*)(out + (size_t)node * HDIM))[lane] = o;
}

// ---------------------------------------------------------------------------
extern "C" void kernel_launch(void* const* d_in, const int* in_sizes, int n_in,
                              void* d_out, int out_size, void* d_ws,
                              size_t ws_size, hipStream_t stream)
{
    (void)in_sizes; (void)n_in; (void)out_size; (void)ws_size;
    const float*     x     = (const float*)d_in[0];
    const long long* ei    = (const long long*)d_in[1];
    const float*     w_pre = (const float*)d_in[2];
    const float*     b_pre = (const float*)d_in[3];
    const float*     gat_w = (const float*)d_in[4];
    const float*     att_s = (const float*)d_in[5];
    const float*     att_d = (const float*)d_in[6];
    const float*     gat_b = (const float*)d_in[7];
    const float*     ln_g  = (const float*)d_in[8];
    const float*     ln_b  = (const float*)d_in[9];
    const float*     mid_w = (const float*)d_in[10];
    const float*     mid_b = (const float*)d_in[11];
    const float*     res_w = (const float*)d_in[12];
    const float*     res_b = (const float*)d_in[13];
    const float*     rln_g = (const float*)d_in[14];
    const float*     rln_b = (const float*)d_in[15];
    const float*     w_out = (const float*)d_in[16];
    const float*     b_out = (const float*)d_in[17];

    float* ws   = (float*)d_ws;
    float* h    = ws;                                    // N*H
    float* prex = h    + (size_t)N_NODES * HDIM;         // N*H
    float* hW   = prex + (size_t)N_NODES * HDIM;         // N*H (reused for o)
    float* agg  = hW   + (size_t)N_NODES * HDIM;         // N*H
    float* mid  = agg  + (size_t)N_NODES * HDIM;         // N*MID
    float* asv  = mid  + (size_t)N_NODES * MIDD;         // N
    float* adv  = asv  + N_NODES;                        // N
    float* zv   = adv  + N_NODES;                        // N
    unsigned* mE = (unsigned*)(zv + N_NODES);            // N
    float* ebuf = (float*)(mE + N_NODES);                // E+N

    const int TOT_E    = N_EDGES + N_NODES;
    const int rowTiles = (N_NODES + 127) / 128;
    const dim3 blk(256);
    const int nodeWaveBlocks = (N_NODES + 7) / 8;

    // h = relu(x @ w_pre + b_pre)
    gemm_f16wmma<DIN, 1><<<dim3(rowTiles, HDIM / 128), blk, 0, stream>>>(
        x, w_pre, b_pre, h, N_NODES, HDIM);
    hipMemcpyAsync(prex, h, sizeof(float) * (size_t)N_NODES * HDIM,
                   hipMemcpyDeviceToDevice, stream);

    for (int l = 0; l < NLAYERS; ++l) {
        // hW = h @ gat_w[l]
        gemm_f16wmma<HDIM, 0><<<dim3(rowTiles, 1), blk, 0, stream>>>(
            h, gat_w + (size_t)l * HDIM * HDIM, nullptr, hW, N_NODES, HDIM);
        alpha_kernel<<<nodeWaveBlocks, blk, 0, stream>>>(
            hW, att_s + l * HDIM, att_d + l * HDIM, asv, adv, N_NODES);
        init_kernel<<<(N_NODES * HDIM + 255) / 256, blk, 0, stream>>>(
            agg, mE, zv, N_NODES);
        edge_pass1<<<(TOT_E + 255) / 256, blk, 0, stream>>>(
            ei, asv, adv, ebuf, mE, TOT_E);
        edge_pass2<<<(TOT_E + 255) / 256, blk, 0, stream>>>(
            ei, ebuf, mE, zv, TOT_E);
        edge_pass3<<<(TOT_E + 7) / 8, blk, 0, stream>>>(
            ei, ebuf, zv, hW, agg, TOT_E);
        post_gat<<<nodeWaveBlocks, blk, 0, stream>>>(
            agg, prex, gat_b + l * HDIM, ln_g, ln_b, h, N_NODES);
        // mid = gelu(h @ mid_w[l] + mid_b[l])
        gemm_f16wmma<HDIM, 2><<<dim3(rowTiles, MIDD / 128), blk, 0, stream>>>(
            h, mid_w + (size_t)l * HDIM * MIDD, mid_b + l * MIDD, mid,
            N_NODES, MIDD);
        // o = mid @ res_w[l] + res_b[l]   (reuse hW buffer)
        gemm_f16wmma<MIDD, 0><<<dim3(rowTiles, 1), blk, 0, stream>>>(
            mid, res_w + (size_t)l * MIDD * HDIM, res_b + l * HDIM, hW,
            N_NODES, HDIM);
        // h = LN(h + o)
        ln_add<<<nodeWaveBlocks, blk, 0, stream>>>(
            h, hW, rln_g + l * HDIM, rln_b + l * HDIM, h, N_NODES);
    }
    // out = relu(h @ w_out + b_out)
    gemm_f16wmma<HDIM, 1><<<dim3(rowTiles, 1), blk, 0, stream>>>(
        h, w_out, b_out, (float*)d_out, N_NODES, DOUT);
}
